// FFSpikingLayer_62723702391149
// MI455X (gfx1250) — compile-verified, matrix operand order checked by reading, not run on hardware
//
#include <hip/hip_runtime.h>
#include <hip/hip_bf16.h>

typedef __attribute__((ext_vector_type(16))) __bf16 v16bf;
typedef __attribute__((ext_vector_type(8)))  float  v8f;

#define T_STEPS 32
#define BATCH   512
#define FEAT    2048                 /* K */
#define HID     2048                 /* N */
#define MDIM    (T_STEPS * BATCH)    /* 16384 rows */

#define BM 128
#define BN 128
#define BK 32
#define LDS_STRIDE 40   /* BK padded: 80B row stride = 16B aligned, bank-conflict free */

__device__ __forceinline__ unsigned short f32_to_bf16_rne(float f) {
  unsigned int u = __float_as_uint(f);
  u += 0x7FFFu + ((u >> 16) & 1u);
  return (unsigned short)(u >> 16);
}

__global__ __launch_bounds__(256) void cvt_f32_bf16(const float* __restrict__ src,
                                                    unsigned short* __restrict__ dst,
                                                    int n4) {
  int i = blockIdx.x * blockDim.x + threadIdx.x;
  if (i >= n4) return;
  float4 f = ((const float4*)src)[i];
  ushort4 o;
  o.x = f32_to_bf16_rne(f.x); o.y = f32_to_bf16_rne(f.y);
  o.z = f32_to_bf16_rne(f.z); o.w = f32_to_bf16_rne(f.w);
  ((ushort4*)dst)[i] = o;
}

__global__ __launch_bounds__(256) void zero_f32(float* __restrict__ p, int n) {
  int i = blockIdx.x * blockDim.x + threadIdx.x;
  if (i < n) p[i] = 0.0f;
}

union Frag16 { uint4 q[2]; v16bf v; };

// Async memory->LDS copy of one 16B chunk (CDNA5 ASYNCcnt path, no VGPR data).
__device__ __forceinline__ void async_b128(unsigned lds_off, const unsigned short* gaddr) {
  asm volatile("global_load_async_to_lds_b128 %0, %1, off"
               :: "v"(lds_off), "v"((unsigned long long)(size_t)gaddr)
               : "memory");
}
__device__ __forceinline__ void wait_async0() {
  asm volatile("s_wait_asynccnt 0x0" ::: "memory");
}

// C[M,N] = A[M,K](bf16) * Bm[N,K](bf16)^T + bias, f32 accum.
// Double-buffered LDS staging via GLOBAL_LOAD_ASYNC_TO_LDS_B128.
// Fused epilogue: per-row sum of squares -> rsq[M] via wave-reduced atomics.
__global__ __launch_bounds__(256) void gemm_bf16_wmma(
    const unsigned short* __restrict__ A,   // [M,K] bf16 (x)
    const unsigned short* __restrict__ Bm,  // [N,K] bf16 (W, row-major H x F)
    const float* __restrict__ bias,         // [N]
    float* __restrict__ C,                  // [M,N] f32 (cur)
    float* __restrict__ rsq)                // [M]   f32 (row sum of squares)
{
  __shared__ alignas(16) unsigned short As[2][BM * LDS_STRIDE];
  __shared__ alignas(16) unsigned short Bs[2][BN * LDS_STRIDE];

  const int tid  = threadIdx.x;
  const int lane = tid & 31;
  const int wave = tid >> 5;          // 8 waves: 4 along M, 2 along N
  const int wM   = (wave >> 1) * 32;
  const int wN   = (wave & 1) * 64;
  const int half = lane >> 4;         // wave32 half
  const int lr   = lane & 15;

  const int bm = blockIdx.x * BM;
  const int bn = blockIdx.y * BN;

  // Per-thread staging coordinates: 512 uint4 per tile, 2 per thread.
  const int r0  = tid >> 2;           // row for chunk 0  (0..63)
  const int r1  = r0 + 64;            // row for chunk 1  (64..127)
  const int c4  = (tid & 3) * 8;      // bf16 column within row

  // Marching global pointers (stride BK elems = 64B per K-step).
  const unsigned short* pa0 = &A [(size_t)(bm + r0) * FEAT + c4];
  const unsigned short* pa1 = &A [(size_t)(bm + r1) * FEAT + c4];
  const unsigned short* pb0 = &Bm[(size_t)(bn + r0) * FEAT + c4];
  const unsigned short* pb1 = &Bm[(size_t)(bn + r1) * FEAT + c4];

  // Loop-invariant LDS destination offsets for both buffers.
  unsigned lA0[2], lA1[2], lB0[2], lB1[2];
#pragma unroll
  for (int buf = 0; buf < 2; ++buf) {
    lA0[buf] = (unsigned)(size_t)&As[buf][r0 * LDS_STRIDE + c4];
    lA1[buf] = (unsigned)(size_t)&As[buf][r1 * LDS_STRIDE + c4];
    lB0[buf] = (unsigned)(size_t)&Bs[buf][r0 * LDS_STRIDE + c4];
    lB1[buf] = (unsigned)(size_t)&Bs[buf][r1 * LDS_STRIDE + c4];
  }

  v8f acc[2][4] = {};

  // Prologue: stage tile k=0 into buffer 0.
  async_b128(lA0[0], pa0);
  async_b128(lA1[0], pa1);
  async_b128(lB0[0], pb0);
  async_b128(lB1[0], pb1);
  pa0 += BK; pa1 += BK; pb0 += BK; pb1 += BK;
  wait_async0();
  __syncthreads();

  const int KSTEPS = FEAT / BK;       // 64
  for (int kt = 0; kt < KSTEPS; ++kt) {
    const int cur = kt & 1;
    const int nxt = cur ^ 1;

    // Prefetch next tile into the other buffer (async engine, ASYNCcnt).
    if (kt + 1 < KSTEPS) {
      async_b128(lA0[nxt], pa0);
      async_b128(lA1[nxt], pa1);
      async_b128(lB0[nxt], pb0);
      async_b128(lB1[nxt], pb1);
      pa0 += BK; pa1 += BK; pb0 += BK; pb1 += BK;
    }

    // Build 16x32 bf16 fragments per ISA layout:
    // lanes 0-15: K {0..7, 16..23}; lanes 16-31: K {8..15, 24..31}
    Frag16 a[2], b[4];
#pragma unroll
    for (int i = 0; i < 2; ++i) {
      const uint4* p = (const uint4*)&As[cur][(wM + i * 16 + lr) * LDS_STRIDE];
      a[i].q[0] = p[half];
      a[i].q[1] = p[2 + half];
    }
#pragma unroll
    for (int j = 0; j < 4; ++j) {
      const uint4* p = (const uint4*)&Bs[cur][(wN + j * 16 + lr) * LDS_STRIDE];
      b[j].q[0] = p[half];
      b[j].q[1] = p[2 + half];
    }
#pragma unroll
    for (int i = 0; i < 2; ++i)
#pragma unroll
      for (int j = 0; j < 4; ++j)
        acc[i][j] = __builtin_amdgcn_wmma_f32_16x16x32_bf16(
            false, a[i].v, false, b[j].v, (short)0, acc[i][j], false, false);

    // Next tile must be resident and everyone must be done reading `cur`
    // before it is overwritten in the following iteration.
    wait_async0();
    __syncthreads();
  }

  // Epilogue: bias add, store cur, per-row sum of squares.
  // C/D layout: lanes 0-15 -> N=lr rows M=e; lanes 16-31 -> N=lr rows M=e+8.
  float sq[2][8];
#pragma unroll
  for (int i = 0; i < 2; ++i)
#pragma unroll
    for (int e = 0; e < 8; ++e) sq[i][e] = 0.0f;

#pragma unroll
  for (int i = 0; i < 2; ++i) {
#pragma unroll
    for (int j = 0; j < 4; ++j) {
      int ng = bn + wN + j * 16 + lr;
      float bv = bias[ng];
#pragma unroll
      for (int e = 0; e < 8; ++e) {
        int mg = bm + wM + i * 16 + half * 8 + e;
        float val = acc[i][j][e] + bv;
        C[(size_t)mg * HID + ng] = val;
        sq[i][e] += val * val;
      }
    }
  }
  // Reduce across the 16 lanes holding different N of the same rows.
#pragma unroll
  for (int i = 0; i < 2; ++i)
#pragma unroll
    for (int e = 0; e < 8; ++e) {
      float s = sq[i][e];
      s += __shfl_xor(s, 1, 16);
      s += __shfl_xor(s, 2, 16);
      s += __shfl_xor(s, 4, 16);
      s += __shfl_xor(s, 8, 16);
      sq[i][e] = s;
    }
  if (lr == 0) {
#pragma unroll
    for (int i = 0; i < 2; ++i)
#pragma unroll
      for (int e = 0; e < 8; ++e)
        atomicAdd(&rsq[bm + wM + i * 16 + half * 8 + e], sq[i][e]);
  }
}

// LIF scan: v <- (v + c)/2 ; s = (v >= 1) ; hard reset v <- v*(1-s)
__global__ __launch_bounds__(256) void lif_scan(
    const float* __restrict__ cur, const float* __restrict__ rsq,
    float* __restrict__ spk, float* __restrict__ count)
{
  int idx = blockIdx.x * blockDim.x + threadIdx.x;   // over BATCH*HID
  int b = idx >> 11;        // / 2048
  int h = idx & 2047;
  float v = 0.0f, cnt = 0.0f;
#pragma unroll
  for (int t = 0; t < T_STEPS; ++t) {
    int row = t * BATCH + b;
    float inv = 1.0f / fmaxf(sqrtf(rsq[row]), 1e-12f);
    float c = cur[(size_t)row * HID + h] * inv;
    v = 0.5f * (v + c);
    float s = (v >= 1.0f) ? 1.0f : 0.0f;
    spk[(size_t)row * HID + h] = s;
    cnt += s;
    v *= (1.0f - s);
  }
  count[idx] = cnt;
}

extern "C" void kernel_launch(void* const* d_in, const int* in_sizes, int n_in,
                              void* d_out, int out_size, void* d_ws, size_t ws_size,
                              hipStream_t stream) {
  const float* x    = (const float*)d_in[0];   // [T,B,F]
  const float* W    = (const float*)d_in[1];   // [H,F]
  const float* bias = (const float*)d_in[2];   // [H]

  // Workspace layout: xb(64MB) | wb(8MB) | cur(128MB) | rsq(64KB)
  char* ws = (char*)d_ws;
  unsigned short* xb = (unsigned short*)ws;
  size_t off = (size_t)MDIM * FEAT * sizeof(unsigned short);
  unsigned short* wb = (unsigned short*)(ws + off);
  off += (size_t)HID * FEAT * sizeof(unsigned short);
  float* cur = (float*)(ws + off);
  off += (size_t)MDIM * HID * sizeof(float);
  float* rsq = (float*)(ws + off);

  float* spk = (float*)d_out;                      // [T,B,H]
  float* cnt = spk + (size_t)MDIM * HID;           // [B,H]

  cvt_f32_bf16<<<(MDIM * FEAT / 4) / 256, 256, 0, stream>>>(x, xb, MDIM * FEAT / 4);
  cvt_f32_bf16<<<(HID * FEAT / 4) / 256, 256, 0, stream>>>(W, wb, HID * FEAT / 4);
  zero_f32<<<MDIM / 256, 256, 0, stream>>>(rsq, MDIM);

  dim3 grid(MDIM / BM, HID / BN);   // 128 x 16 workgroups
  gemm_bf16_wmma<<<grid, 256, 0, stream>>>(xb, wb, bias, cur, rsq);

  lif_scan<<<(BATCH * HID) / 256, 256, 0, stream>>>(cur, rsq, spk, cnt);
}
